// PrefixAttention_57011395887700
// MI455X (gfx1250) — matrix operand order repaired and checked by reading, not compiled, not run on hardware
//
#include <hip/hip_runtime.h>
#include <hip/hip_bf16.h>

typedef __bf16 bf16;
typedef __bf16 v16bf __attribute__((ext_vector_type(16)));
typedef __bf16 v8bf  __attribute__((ext_vector_type(8)));
typedef float  v8f   __attribute__((ext_vector_type(8)));

#define BB   4
#define NN   2048
#define EE   1024
#define HH   16
#define PP   16
#define DD   64
#define MM   2064      // P + N
#define MPAD 2080      // padded to multiple of 32

static __device__ __forceinline__ v16bf cat8(v8bf lo, v8bf hi) {
  return __builtin_shufflevector(lo, hi, 0,1,2,3,4,5,6,7,8,9,10,11,12,13,14,15);
}

static __device__ __forceinline__ v8f wmma_bf16(v16bf a, v16bf b, v8f c) {
  // D = A(16x32) * B(32x16) + C, f32 accum
  return __builtin_amdgcn_wmma_f32_16x16x32_bf16(false, a, false, b, (short)0, c, false, false);
}

static __device__ __forceinline__ float rmax16(float v) {
  v = fmaxf(v, __shfl_xor(v, 1));
  v = fmaxf(v, __shfl_xor(v, 2));
  v = fmaxf(v, __shfl_xor(v, 4));
  v = fmaxf(v, __shfl_xor(v, 8));
  return v;
}
static __device__ __forceinline__ float rsum16(float v) {
  v += __shfl_xor(v, 1);
  v += __shfl_xor(v, 2);
  v += __shfl_xor(v, 4);
  v += __shfl_xor(v, 8);
  return v;
}

// Async global -> LDS copy of 16 bytes (per lane), tracked by ASYNCcnt.
// ldsoff = LDS byte offset (generic LDS pointer truncated to 32 bits),
// gaddr  = 64-bit global byte address.
static __device__ __forceinline__ void async_copy_b128(unsigned ldsoff, unsigned long long gaddr) {
  asm volatile("global_load_async_to_lds_b128 %0, %1, off"
               :: "v"(ldsoff), "v"(gaddr) : "memory");
}
static __device__ __forceinline__ void wait_asynccnt0() {
  asm volatile("s_wait_asynccnt 0" ::: "memory");
}
static __device__ __forceinline__ unsigned lds_off(const void* p) {
  return (unsigned)(unsigned long long)(uintptr_t)p;   // LDS aperture: addr[31:0] = LDS offset
}

// ---------------- f32 -> bf16 convert ----------------
__global__ __launch_bounds__(256) void cvt_kernel(const float* __restrict__ in,
                                                  bf16* __restrict__ out, int n) {
  int i = blockIdx.x * 256 + threadIdx.x;
  if (i < n) out[i] = (bf16)in[i];
}

// ------------- K/V prefix init + tail zero-pad -------------
// kall: (B*H, MPAD, D) bf16 ; vt: (B*H, D, MPAD) bf16 (V transposed)
__global__ __launch_bounds__(256) void prefix_init_kernel(const float* __restrict__ kp,
                                                          const float* __restrict__ vp,
                                                          bf16* __restrict__ kall,
                                                          bf16* __restrict__ vt) {
  int i = blockIdx.x * 256 + threadIdx.x;        // B*H*32*D = 131072 threads
  int d   = i & 63;
  int row = (i >> 6) & 31;
  int bh  = i >> 11;
  if (row < PP) {
    float kv = kp[((size_t)bh * PP + row) * DD + d];
    float vv = vp[((size_t)bh * PP + row) * DD + d];
    kall[((size_t)bh * MPAD + row) * DD + d] = (bf16)kv;
    vt[((size_t)bh * DD + d) * MPAD + row]   = (bf16)vv;
  } else {
    int col = MM + (row - PP);                   // 2064..2079 pad
    kall[((size_t)bh * MPAD + col) * DD + d] = (bf16)0.0f;
    vt[((size_t)bh * DD + d) * MPAD + col]   = (bf16)0.0f;
  }
}

// ---------------- WMMA GEMM: Y[m,n] = sum_k A[m,k]*W[n,k] + bias[n] ----------------
// A: (8192,1024) bf16 row-major, W: (1024,1024) bf16 row-major.
// MODE 0: bf16 plain (m,e)        (Q)
// MODE 1: bf16 -> kall layout     (K)
// MODE 2: bf16 -> vt  layout      (V, transposed)
// MODE 3: f32  plain (m,e)        (final projection -> d_out)
template <int MODE>
__global__ __launch_bounds__(256) void gemm_bf16_kernel(const bf16* __restrict__ A,
                                                        const bf16* __restrict__ W,
                                                        const float* __restrict__ bias,
                                                        void* __restrict__ out) {
  const int lane = threadIdx.x & 31;
  const int wave = threadIdx.x >> 5;
  const int gw   = blockIdx.x * 8 + wave;     // 0..8191
  const int mt   = gw >> 4;                   // 16-row tile index, 0..511
  const int nt   = (gw & 15) * 64;            // 64-col stripe base
  const int half = lane >> 4;
  const int l15  = lane & 15;

  const int arow = mt * 16 + l15;             // A-fragment row for this lane
  const bf16* Ap = A + (size_t)arow * EE;
  const bf16* Wp = W + (size_t)(nt + l15) * EE;

  v8f c0 = {}, c1 = {}, c2 = {}, c3 = {};

  for (int k0 = 0; k0 < EE; k0 += 32) {
    // Prefetch the next k-step of the streamed operands (global_prefetch_b8).
    if (k0 + 32 < EE) {
      __builtin_prefetch(Ap + k0 + 32, 0, 0);
      __builtin_prefetch(Wp + k0 + 32, 0, 0);
    }
    // A fragment: lane holds row `arow`, k = k0 + half*16 + 0..15
    const bf16* ap = Ap + k0 + half * 16;
    v16bf a = *(const v16bf*)(ap);
    // B fragments: column n -> W row n, k = k0 + half*16 + 0..15 (contiguous 32B)
    const bf16* wp = Wp + k0 + half * 16;
    c0 = wmma_bf16(a, *(const v16bf*)(wp),                   c0);
    c1 = wmma_bf16(a, *(const v16bf*)(wp + 16 * (size_t)EE), c1);
    c2 = wmma_bf16(a, *(const v16bf*)(wp + 32 * (size_t)EE), c2);
    c3 = wmma_bf16(a, *(const v16bf*)(wp + 48 * (size_t)EE), c3);
  }

  v8f cs[4] = {c0, c1, c2, c3};
#pragma unroll
  for (int s = 0; s < 4; ++s) {
    const int col = nt + s * 16 + l15;          // output feature e
    const float bv = bias[col];
#pragma unroll
    for (int r = 0; r < 8; ++r) {
      const int row = mt * 16 + half * 8 + r;   // global m
      const float val = cs[s][r] + bv;
      if (MODE == 0) {
        ((bf16*)out)[(size_t)row * EE + col] = (bf16)val;
      } else if (MODE == 1) {
        const int b = row >> 11, n = row & (NN - 1);
        const int h = col >> 6,  d = col & 63;
        ((bf16*)out)[(((size_t)(b * HH + h)) * MPAD + PP + n) * DD + d] = (bf16)val;
      } else if (MODE == 2) {
        const int b = row >> 11, n = row & (NN - 1);
        const int h = col >> 6,  d = col & 63;
        ((bf16*)out)[(((size_t)(b * HH + h)) * DD + d) * MPAD + PP + n] = (bf16)val;
      } else {
        ((float*)out)[(size_t)row * EE + col] = val;
      }
    }
  }
}

// ---------------- Flash attention ----------------
// 8 waves per block share one (b,h); K/V tiles are staged into LDS once per
// block per 32-key step with async global->LDS copies (ASYNCcnt), then all
// waves feed WMMA B-fragments from LDS.
__global__ __launch_bounds__(256) void attn_kernel(const bf16* __restrict__ q,
                                                   const bf16* __restrict__ kall,
                                                   const bf16* __restrict__ vt,
                                                   bf16* __restrict__ o) {
  __shared__ __align__(32) bf16 ktile[32][64];        // 4 KB: [key][d]
  __shared__ __align__(32) bf16 vtile[64][32];        // 4 KB: [d][key]
  __shared__ __align__(32) bf16 ptile[8][16][32];     // 8 KB: per-wave P transpose

  const int tid  = threadIdx.x;
  const int lane = tid & 31;
  const int wave = tid >> 5;
  const int half = lane >> 4;
  const int l15  = lane & 15;
  const int gw   = blockIdx.x * 8 + wave;     // 0..8191
  const int bh   = gw >> 7;                   // 0..63  (b*H + h)
  const int qt   = gw & 127;                  // 16-query tile within (b,h)
  const int b    = bh >> 4;
  const int h    = bh & 15;

  // Q fragments (two 32-wide d chunks): lane row = qt*16 + l15
  const bf16* qrow = q + ((size_t)(b * NN + qt * 16 + l15)) * EE + h * DD;
  v16bf qa[2];
#pragma unroll
  for (int c = 0; c < 2; ++c) {
    const bf16* p = qrow + c * 32 + half * 16;
    qa[c] = *(const v16bf*)(p);
  }

  const bf16* kbase = kall + (size_t)bh * MPAD * DD;
  const bf16* vbase = vt   + (size_t)bh * DD * MPAD;

  // Async-copy source/dest (16B per thread per tile):
  // K tile is one contiguous 4KB block (rows contiguous).
  const unsigned k_ldst = lds_off(ktile) + (unsigned)tid * 16;
  // V tile: 64 rows (stride MPAD elems) x 64B; thread -> (row = tid>>2, chunk = tid&3)
  const int vrow = tid >> 2, vch = tid & 3;
  const unsigned v_ldst = lds_off(vtile) + (unsigned)(vrow * 64 + vch * 16);

  float m_i[8], l_i[8];
#pragma unroll
  for (int r = 0; r < 8; ++r) { m_i[r] = -1e30f; l_i[r] = 0.0f; }
  v8f acc[4] = {{}, {}, {}, {}};

  const float scale = 0.125f;                 // 1/sqrt(64)

  for (int kb = 0; kb < MPAD; kb += 32) {
    // ---- stage K (32x64) and V^T (64x32) tiles into LDS ----
    async_copy_b128(k_ldst, (unsigned long long)(uintptr_t)
                    ((const char*)kbase + (size_t)kb * DD * 2 + (size_t)tid * 16));
    async_copy_b128(v_ldst, (unsigned long long)(uintptr_t)
                    ((const char*)vbase + ((size_t)vrow * MPAD + kb) * 2 + (size_t)vch * 16));
    wait_asynccnt0();
    __syncthreads();

    // S tiles (16 queries x 32 keys) via 4 WMMAs, B-fragments from LDS
    v8f s0 = {}, s1 = {};
#pragma unroll
    for (int c = 0; c < 2; ++c) {
      s0 = wmma_bf16(qa[c], *(const v16bf*)&ktile[l15][c * 32 + half * 16], s0);
      s1 = wmma_bf16(qa[c], *(const v16bf*)&ktile[16 + l15][c * 32 + half * 16], s1);
    }
    const bool msk0 = (kb + l15)      >= MM;
    const bool msk1 = (kb + 16 + l15) >= MM;

    float f[8], p0[8], p1[8];
#pragma unroll
    for (int r = 0; r < 8; ++r) {
      const float a0 = msk0 ? -1e30f : s0[r] * scale;
      const float a1 = msk1 ? -1e30f : s1[r] * scale;
      const float tmax = rmax16(fmaxf(a0, a1));
      const float mn = fmaxf(m_i[r], tmax);
      f[r] = __expf(m_i[r] - mn);
      m_i[r] = mn;
      p0[r] = __expf(a0 - mn);
      p1[r] = __expf(a1 - mn);
      const float rs = rsum16(p0[r] + p1[r]);
      l_i[r] = l_i[r] * f[r] + rs;
    }
#pragma unroll
    for (int ch = 0; ch < 4; ++ch)
#pragma unroll
      for (int r = 0; r < 8; ++r) acc[ch][r] *= f[r];

    // P (16x32) C-layout -> LDS -> A-layout bf16 fragment (wave-local)
#pragma unroll
    for (int r = 0; r < 8; ++r) {
      ptile[wave][half * 8 + r][l15]      = (bf16)p0[r];
      ptile[wave][half * 8 + r][16 + l15] = (bf16)p1[r];
    }
    asm volatile("s_wait_dscnt 0" ::: "memory");
    v16bf pa = *(const v16bf*)&ptile[wave][l15][half * 16];

    // O += P * V : B fragment from transposed-V tile in LDS
#pragma unroll
    for (int ch = 0; ch < 4; ++ch) {
      acc[ch] = wmma_bf16(pa, *(const v16bf*)&vtile[ch * 16 + l15][half * 16], acc[ch]);
    }
    __syncthreads();   // all waves done reading K/V tiles before next overwrite
  }

  float rl[8];
#pragma unroll
  for (int r = 0; r < 8; ++r) rl[r] = 1.0f / l_i[r];
#pragma unroll
  for (int ch = 0; ch < 4; ++ch)
#pragma unroll
    for (int r = 0; r < 8; ++r) {
      const int row = qt * 16 + half * 8 + r;
      o[((size_t)(b * NN + row)) * EE + h * DD + ch * 16 + l15] = (bf16)(acc[ch][r] * rl[r]);
    }
}

extern "C" void kernel_launch(void* const* d_in, const int* in_sizes, int n_in,
                              void* d_out, int out_size, void* d_ws, size_t ws_size,
                              hipStream_t stream) {
  const float* x  = (const float*)d_in[0];
  const float* kp = (const float*)d_in[1];
  const float* vp = (const float*)d_in[2];
  const float* wq = (const float*)d_in[3];
  const float* bq = (const float*)d_in[4];
  const float* wk = (const float*)d_in[5];
  const float* bk = (const float*)d_in[6];
  const float* wv = (const float*)d_in[7];
  const float* bv = (const float*)d_in[8];
  const float* wo = (const float*)d_in[9];
  const float* bo = (const float*)d_in[10];

  char* ws = (char*)d_ws;
  bf16* xb   = (bf16*)(ws);                  // 8192*1024*2       = 16,777,216
  bf16* wqb  = (bf16*)(ws + 16777216);       // 1024*1024*2 each
  bf16* wkb  = (bf16*)(ws + 18874368);
  bf16* wvb  = (bf16*)(ws + 20971520);
  bf16* wob  = (bf16*)(ws + 23068672);
  bf16* qb   = (bf16*)(ws + 25165824);       // 16,777,216
  bf16* kall = (bf16*)(ws + 41943040);       // 4*16*2080*64*2    = 17,039,360
  bf16* vtp  = (bf16*)(ws + 58982400);       // 17,039,360
  bf16* ob   = (bf16*)(ws + 76021760);       // 16,777,216  (end ~88.5 MB)

  cvt_kernel<<<(BB * NN * EE + 255) / 256, 256, 0, stream>>>(x,  xb,  BB * NN * EE);
  cvt_kernel<<<(EE * EE + 255) / 256,     256, 0, stream>>>(wq, wqb, EE * EE);
  cvt_kernel<<<(EE * EE + 255) / 256,     256, 0, stream>>>(wk, wkb, EE * EE);
  cvt_kernel<<<(EE * EE + 255) / 256,     256, 0, stream>>>(wv, wvb, EE * EE);
  cvt_kernel<<<(EE * EE + 255) / 256,     256, 0, stream>>>(wo, wob, EE * EE);

  prefix_init_kernel<<<(BB * HH * 32 * DD) / 256, 256, 0, stream>>>(kp, vp, kall, vtp);

  // 8192 waves = 1024 blocks x 8 waves per GEMM / attention launch
  gemm_bf16_kernel<0><<<1024, 256, 0, stream>>>(xb, wqb, bq, qb);
  gemm_bf16_kernel<1><<<1024, 256, 0, stream>>>(xb, wkb, bk, kall);
  gemm_bf16_kernel<2><<<1024, 256, 0, stream>>>(xb, wvb, bv, vtp);

  attn_kernel<<<1024, 256, 0, stream>>>(qb, kall, vtp, ob);

  gemm_bf16_kernel<3><<<1024, 256, 0, stream>>>(ob, wob, bo, d_out);
}
